// MaskedAttentionPooling_88888643158750
// MI455X (gfx1250) — compile-verified, matrix operand order checked by reading, not run on hardware
//
#include <hip/hip_runtime.h>
#include <stdint.h>

// Problem sizes (fixed by the reference)
#define B_ 64
#define T_ 2048
#define D_ 512

#define MTILE 64                  // rows of x per workgroup (4 waves x 16)
#define NWAVES 4
#define NTHREADS 128
#define LDS_PAD 8                 // pad (bf16 elems) to kill LDS bank conflicts
#define LDS_STRIDE (D_ + LDS_PAD) // 520 halves per row = 1040 B (16B aligned)

typedef __attribute__((ext_vector_type(16))) __bf16 v16bf;
typedef __attribute__((ext_vector_type(8)))  float  v8f;

union Frag {
  uint4 u[2];   // 32 bytes
  v16bf v;      // 16 x bf16
};

__device__ __forceinline__ unsigned short f2bf(float f) {
  unsigned int u = __float_as_uint(f);
  u += 0x7FFFu + ((u >> 16) & 1u);   // round-to-nearest-even
  return (unsigned short)(u >> 16);
}

// ---------------------------------------------------------------------------
// Kernel 1: WqT[n][k] = bf16(Wq[k][n])  (h = x @ Wq, so B[k][n] = Wq[k*D+n])
// ---------------------------------------------------------------------------
__global__ void wq_to_bf16T(const float* __restrict__ Wq,
                            unsigned short* __restrict__ WqT) {
  int idx = blockIdx.x * blockDim.x + threadIdx.x;  // 0 .. D*D-1
  int k = idx >> 9;        // row of Wq (coalesced read along n)
  int n = idx & (D_ - 1);
  WqT[n * D_ + k] = f2bf(Wq[idx]);
}

// ---------------------------------------------------------------------------
// Async stage of one 16 x 512 bf16 B tile (WqT rows nt*16..+15) into LDS.
// Uses CDNA5 GLOBAL_LOAD_ASYNC_TO_LDS_B128 (GVS mode): no VGPR round-trip,
// tracked on ASYNCcnt -> overlaps with WMMA compute of the previous tile.
// ---------------------------------------------------------------------------
__device__ __forceinline__ void stageB_async(const unsigned short* __restrict__ WqT,
                                             unsigned short* ldsB, int nt, int tid) {
  // flat shared pointer: low 32 bits are the LDS byte offset
  unsigned ldsBase = (unsigned)(uintptr_t)ldsB;
  const unsigned short* gbase = WqT + nt * 16 * D_;
  #pragma unroll
  for (int j = 0; j < 8; ++j) {
    int idx = tid + NTHREADS * j;        // 1024 x uint4 total; 64 uint4 per row
    int r   = idx >> 6;
    int c8  = idx & 63;
    unsigned loff = ldsBase + (unsigned)((r * LDS_STRIDE + c8 * 8) * 2);
    unsigned goff = (unsigned)((r * D_ + c8 * 8) * 2);
    asm volatile("global_load_async_to_lds_b128 %0, %1, %2"
                 :: "v"(loff), "v"(goff), "s"(gbase)
                 : "memory");
  }
}

__device__ __forceinline__ void wait_async0() {
  asm volatile("s_wait_asynccnt 0x0" ::: "memory");
}

__device__ __forceinline__ void loadFragA(Frag& f, const unsigned short* ldsA,
                                          int mrow, int k0, int lh) {
  f.u[0] = *((const uint4*)&ldsA[mrow * LDS_STRIDE + k0 + lh * 8]);
  f.u[1] = *((const uint4*)&ldsA[mrow * LDS_STRIDE + k0 + 16 + lh * 8]);
}
__device__ __forceinline__ void loadFragB(Frag& f, const unsigned short* ldsB,
                                          int nrow, int k0, int lh) {
  f.u[0] = *((const uint4*)&ldsB[nrow * LDS_STRIDE + k0 + lh * 8]);
  f.u[1] = *((const uint4*)&ldsB[nrow * LDS_STRIDE + k0 + 16 + lh * 8]);
}

// ---------------------------------------------------------------------------
// Kernel 2: fused logits[r] = sum_n tanh( (x @ Wq)[r,n] ) * ws[n]
// One WG = 128 threads = 4 wave32, 64 rows of x. bf16 WMMA, f32 accum.
// B tile double-buffered via async-to-LDS; fragments double-buffered in K.
// ---------------------------------------------------------------------------
__global__ void logits_wmma(const float* __restrict__ x,
                            const unsigned short* __restrict__ WqT,
                            const float* __restrict__ ws,
                            float* __restrict__ logits) {
  extern __shared__ unsigned short smem[];
  unsigned short* ldsA  = smem;                        // MTILE x LDS_STRIDE
  unsigned short* ldsB0 = smem + MTILE * LDS_STRIDE;   // 16 x LDS_STRIDE
  unsigned short* ldsB1 = ldsB0 + 16 * LDS_STRIDE;     // 16 x LDS_STRIDE

  const int tid  = threadIdx.x;
  const int lane = tid & 31;
  const int wave = tid >> 5;
  const int lm   = lane & 15;   // position within half-wave (= n column)
  const int lh   = lane >> 4;   // which half-wave (K / M split)
  const long rowBase = (long)blockIdx.x * MTILE;

  // kick off async staging of the first B tile; overlaps the A staging below
  stageB_async(WqT, ldsB0, 0, tid);

  // ---- stage x tile (fp32 global -> bf16 LDS), batched 4 loads per wait ----
  const float4* xg = (const float4*)(x + rowBase * D_);
  #pragma unroll 1
  for (int it = 0; it < (MTILE * (D_ / 4)) / (NTHREADS * 4); ++it) { // 16 iters
    float4 f[4];
    int base = tid + it * (NTHREADS * 4);
    #pragma unroll
    for (int j = 0; j < 4; ++j) f[j] = xg[base + NTHREADS * j];
    #pragma unroll
    for (int j = 0; j < 4; ++j) {
      int idx = base + NTHREADS * j;
      int r   = idx >> 7;          // D_/4 = 128 float4 per row
      int c4  = idx & 127;
      unsigned lo = (unsigned)f2bf(f[j].x) | ((unsigned)f2bf(f[j].y) << 16);
      unsigned hi = (unsigned)f2bf(f[j].z) | ((unsigned)f2bf(f[j].w) << 16);
      *((uint2*)&ldsA[r * LDS_STRIDE + c4 * 4]) = make_uint2(lo, hi);
    }
  }

  wait_async0();
  __syncthreads();

  float part[8] = {0.f, 0.f, 0.f, 0.f, 0.f, 0.f, 0.f, 0.f};
  const int mrow = wave * 16 + lm;  // local A row this lane supplies

  for (int nt = 0; nt < D_ / 16; ++nt) {
    unsigned short* bcur = (nt & 1) ? ldsB1 : ldsB0;
    // prefetch next B tile into the other buffer while we compute this one
    if (nt < D_ / 16 - 1)
      stageB_async(WqT, (nt & 1) ? ldsB0 : ldsB1, nt + 1, tid);

    v8f c = {0.f, 0.f, 0.f, 0.f, 0.f, 0.f, 0.f, 0.f};
    Frag a2[2], b2[2];
    loadFragA(a2[0], ldsA, mrow, 0, lh);
    loadFragB(b2[0], bcur, lm, 0, lh);
    #pragma unroll
    for (int kt = 0; kt < D_ / 32; ++kt) {
      const int cu = kt & 1;
      if (kt < D_ / 32 - 1) {       // software pipeline: next frags before wmma
        loadFragA(a2[cu ^ 1], ldsA, mrow, (kt + 1) * 32, lh);
        loadFragB(b2[cu ^ 1], bcur, lm, (kt + 1) * 32, lh);
      }
      c = __builtin_amdgcn_wmma_f32_16x16x32_bf16(
              /*neg_a=*/false, a2[cu].v, /*neg_b=*/false, b2[cu].v,
              /*c_mod=*/(short)0, c, /*reuse_a=*/false, /*reuse_b=*/false);
    }

    // C/D layout: n = lane&15, m = v + 8*lh  -> fold tanh * ws[n] now
    float wsv = ws[nt * 16 + lm];
    #pragma unroll
    for (int v = 0; v < 8; ++v) part[v] += tanhf(c[v]) * wsv;

    wait_async0();     // next B tile fully in LDS
    __syncthreads();   // and everyone done reading the old one
  }

  // reduce over the 16 n-lanes of each half-wave
  #pragma unroll
  for (int v = 0; v < 8; ++v) {
    float p = part[v];
    p += __shfl_xor(p, 8, 16);
    p += __shfl_xor(p, 4, 16);
    p += __shfl_xor(p, 2, 16);
    p += __shfl_xor(p, 1, 16);
    part[v] = p;
  }
  if (lm == 0) {
    #pragma unroll
    for (int v = 0; v < 8; ++v)
      logits[rowBase + wave * 16 + lh * 8 + v] = part[v];
  }
}

// ---------------------------------------------------------------------------
// Kernel 3: masked softmax over T per batch row
// ---------------------------------------------------------------------------
__global__ void masked_softmax(const float* __restrict__ logits,
                               const unsigned char* __restrict__ mask,
                               float* __restrict__ w) {
  const int b   = blockIdx.x;
  const int tid = threadIdx.x;
  __shared__ float red[256];

  float v[8];
  float lmax = -3.4e38f;
  #pragma unroll
  for (int j = 0; j < 8; ++j) {
    int t = j * 256 + tid;
    float lg = logits[b * T_ + t];
    if (!mask[b * T_ + t]) lg = -1e9f;
    v[j] = lg;
    lmax = fmaxf(lmax, lg);
  }
  red[tid] = lmax;
  __syncthreads();
  for (int s = 128; s > 0; s >>= 1) {
    if (tid < s) red[tid] = fmaxf(red[tid], red[tid + s]);
    __syncthreads();
  }
  float m = red[0];
  __syncthreads();

  float lsum = 0.f;
  #pragma unroll
  for (int j = 0; j < 8; ++j) { v[j] = __expf(v[j] - m); lsum += v[j]; }
  red[tid] = lsum;
  __syncthreads();
  for (int s = 128; s > 0; s >>= 1) {
    if (tid < s) red[tid] += red[tid + s];
    __syncthreads();
  }
  float inv = 1.0f / red[0];
  #pragma unroll
  for (int j = 0; j < 8; ++j) w[b * T_ + j * 256 + tid] = v[j] * inv;
}

// ---------------------------------------------------------------------------
// Kernel 4: z[b,d] = sum_t w[b,t] * x[b,t,d]   (memory-bound, coalesced)
// ---------------------------------------------------------------------------
__global__ void weighted_pool(const float* __restrict__ x,
                              const float* __restrict__ w,
                              float* __restrict__ z) {
  const int b = blockIdx.y;
  const int d = blockIdx.x * 256 + threadIdx.x;
  __shared__ float wsh[256];
  const float* xb = x + (size_t)b * T_ * D_;
  const float* wb = w + (size_t)b * T_;
  float acc = 0.f;
  for (int tc = 0; tc < T_; tc += 256) {
    __syncthreads();
    wsh[threadIdx.x] = wb[tc + threadIdx.x];
    __syncthreads();
    #pragma unroll 8
    for (int tt = 0; tt < 256; ++tt)
      acc += wsh[tt] * xb[(size_t)(tc + tt) * D_ + d];
  }
  z[b * D_ + d] = acc;
}

// ---------------------------------------------------------------------------
extern "C" void kernel_launch(void* const* d_in, const int* in_sizes, int n_in,
                              void* d_out, int out_size, void* d_ws, size_t ws_size,
                              hipStream_t stream) {
  (void)in_sizes; (void)n_in; (void)out_size; (void)ws_size;

  const float*         x    = (const float*)d_in[0];          // [B,T,D] f32
  const unsigned char* mask = (const unsigned char*)d_in[1];  // [B,T] bool
  const float*         Wq   = (const float*)d_in[2];          // [D,D] f32
  const float*         ws   = (const float*)d_in[3];          // [D,1] f32
  float*               z    = (float*)d_out;                  // [B,D] f32

  // scratch layout: logits | softmax weights | WqT(bf16)
  float*          logits = (float*)d_ws;                       // B*T floats
  float*          wgt    = logits + (size_t)B_ * T_;           // B*T floats
  unsigned short* WqT    = (unsigned short*)(wgt + (size_t)B_ * T_); // D*D bf16

  wq_to_bf16T<<<(D_ * D_) / 256, 256, 0, stream>>>(Wq, WqT);

  size_t smem = (size_t)(MTILE + 32) * LDS_STRIDE * sizeof(unsigned short); // ~100 KB
  logits_wmma<<<(B_ * T_) / MTILE, NTHREADS, smem, stream>>>(x, WqT, ws, logits);

  masked_softmax<<<B_, 256, 0, stream>>>(logits, mask, wgt);

  weighted_pool<<<dim3(D_ / 256, B_), 256, 0, stream>>>(x, wgt, z);
}
